// FFTConv_12025908428820
// MI455X (gfx1250) — matrix operand order repaired and checked by reading, not compiled, not run on hardware
//
#include <hip/hip_runtime.h>

typedef __attribute__((ext_vector_type(16))) _Float16 v16h;
typedef __attribute__((ext_vector_type(8)))  float    v8f;

#define HH   112
#define WW   112
#define ICN  128
#define OCN  256
#define RSQ2 0.70710678118654752f

// cos/sin(2*pi*k/8) for the (cold) T-packing kernel
__device__ __forceinline__ float c8f(int k) {
  const float t[8] = {1.f, RSQ2, 0.f, -RSQ2, -1.f, -RSQ2, 0.f, RSQ2};
  return t[k & 7];
}
__device__ __forceinline__ float s8f(int k) {
  const float t[8] = {0.f, RSQ2, 1.f, RSQ2, 0.f, -RSQ2, -1.f, -RSQ2};
  return t[k & 7];
}

// ---------------------------------------------------------------------------
// Pack wreal/wimag (IC,OC,8,8) fp32 -> f16 B-operand tiles in WMMA lane order.
// Layout: [f=64][ri=2][kc=4][nt=16][512 f16]; element q: lane=q>>4, e=q&15,
// vg=e>>1, hf=e&1;  n = lane&15,  k = (lane>>4)*16 + vg*2 + hf   (B 32x16).
// ---------------------------------------------------------------------------
__global__ __launch_bounds__(256) void pack_w(const float* __restrict__ wr,
                                              const float* __restrict__ wi,
                                              _Float16* __restrict__ wpk) {
  const unsigned idx = blockIdx.x * 256u + threadIdx.x;   // 0 .. 4194303
  const int q  = idx & 511;
  const int nt = (idx >> 9) & 15;
  const int kc = (idx >> 13) & 3;
  const int ri = (idx >> 15) & 1;
  const int f  = idx >> 16;            // 0..63
  const int lane = q >> 4;
  const int e    = q & 15;
  const int n = lane & 15;
  const int k = (lane >> 4) * 16 + (e >> 1) * 2 + (e & 1);
  const int ic = kc * 32 + k;
  const int oc = nt * 16 + n;
  const float* src = ri ? wi : wr;
  wpk[idx] = (_Float16)src[((size_t)ic * OCN + oc) * 64 + f];
}

// ---------------------------------------------------------------------------
// Pack iFFT twiddle GEMM operand T[k=2*fl+ri][s] (K=128, N=16) as f16 B tiles:
// 4 chunks of [512 f16].  T[2fl+0][s] = cos(th), T[2fl+1][s] = -sin(th),
// th = 2pi*(u*(sy+2) + v*(sx+2))/8,  fl = u*8+v,  s = sy*4+sx.
// ---------------------------------------------------------------------------
__global__ __launch_bounds__(256) void pack_t(_Float16* __restrict__ wt) {
  const int idx = blockIdx.x * 256 + threadIdx.x;   // 0..2047
  const int q  = idx & 511;
  const int kq = idx >> 9;
  const int lane = q >> 4;
  const int e    = q & 15;
  const int s  = lane & 15;
  const int kl = (lane >> 4) * 16 + (e >> 1) * 2 + (e & 1);
  const int k  = kq * 32 + kl;
  const int fl = k >> 1, ri = k & 1;
  const int u = fl >> 3, v = fl & 7;
  const int pidx = (u * ((s >> 2) + 2) + v * ((s & 3) + 2)) & 7;
  wt[idx] = (_Float16)(ri ? -s8f(pidx) : c8f(pidx));
}

// ---------------------------------------------------------------------------
// Factored 8-point column DFT (input real), writes 8 freqs of row u.
// ---------------------------------------------------------------------------
__device__ __forceinline__ void colDFT_r(int u, const float g[6],
                                         _Float16* Alds, int eoff) {
  const float S = g[0]+g[1]+g[2]+g[3]+g[4]+g[5];
  const float T = g[0]-g[1]+g[2]-g[3]+g[4]-g[5];
  const float D = g[0]-g[2]+g[4];
  const float E = g[1]-g[3]+g[5];
  const float A = g[0]-g[4];
  const float B = g[1]-g[3]-g[5];
  const float C = g[1]+g[3]-g[5];
  const float rB = RSQ2*B, rC = RSQ2*C;
  float Pr[8], Pi[8];
  Pr[0]=S;     Pi[0]=0.f;
  Pr[1]=A+rB;  Pi[1]=-(g[2]+rC);
  Pr[2]=D;     Pi[2]=-E;
  Pr[3]=A-rB;  Pi[3]=g[2]-rC;
  Pr[4]=T;     Pi[4]=0.f;
  Pr[5]=A-rB;  Pi[5]=-(g[2]-rC);
  Pr[6]=D;     Pi[6]=E;
  Pr[7]=A+rB;  Pi[7]=g[2]+rC;
  #pragma unroll
  for (int v = 0; v < 8; ++v) {
    const int fl = u * 8 + v;
    Alds[(fl*2+0)*512 + eoff] = (_Float16)Pr[v];
    Alds[(fl*2+1)*512 + eoff] = (_Float16)Pi[v];
  }
}

// Factored 8-point column DFT, complex input row u (1..3); also emits row 8-u
// via P[8-u][v] = conj(P[u][(8-v)&7]).
__device__ __forceinline__ void colDFT_c(int u, const float gr[6], const float gi[6],
                                         _Float16* Alds, int eoff) {
  const float Sr = gr[0]+gr[1]+gr[2]+gr[3]+gr[4]+gr[5];
  const float Si = gi[0]+gi[1]+gi[2]+gi[3]+gi[4]+gi[5];
  const float Tr = gr[0]-gr[1]+gr[2]-gr[3]+gr[4]-gr[5];
  const float Ti = gi[0]-gi[1]+gi[2]-gi[3]+gi[4]-gi[5];
  const float Dr = gr[0]-gr[2]+gr[4], Di = gi[0]-gi[2]+gi[4];
  const float Er = gr[1]-gr[3]+gr[5], Ei = gi[1]-gi[3]+gi[5];
  const float Ar = gr[0]-gr[4],       Ai = gi[0]-gi[4];
  const float Br = gr[1]-gr[3]-gr[5], Bi = gi[1]-gi[3]-gi[5];
  const float Cr = gr[1]+gr[3]-gr[5], Ci = gi[1]+gi[3]-gi[5];
  const float rBr = RSQ2*Br, rBi = RSQ2*Bi, rCr = RSQ2*Cr, rCi = RSQ2*Ci;
  const float X1r = Ar+rBr,    X1i = Ai+rBi;      // A + rB
  const float Y1r = gr[2]+rCr, Y1i = gi[2]+rCi;   // g2 + rC
  const float X3r = Ar-rBr,    X3i = Ai-rBi;      // A - rB
  const float Y3r = gr[2]-rCr, Y3i = gi[2]-rCi;   // g2 - rC
  float Pr[8], Pi[8];
  Pr[0]=Sr;       Pi[0]=Si;
  Pr[4]=Tr;       Pi[4]=Ti;
  Pr[2]=Dr+Ei;    Pi[2]=Di-Er;    // D - iE
  Pr[6]=Dr-Ei;    Pi[6]=Di+Er;    // D + iE
  Pr[1]=X1r+Y1i;  Pi[1]=X1i-Y1r;  // X1 - iY1
  Pr[7]=X1r-Y1i;  Pi[7]=X1i+Y1r;  // X1 + iY1
  Pr[3]=X3r-Y3i;  Pi[3]=X3i+Y3r;  // X3 + iY3
  Pr[5]=X3r+Y3i;  Pi[5]=X3i-Y3r;  // X3 - iY3
  #pragma unroll
  for (int v = 0; v < 8; ++v) {
    const int fl = u * 8 + v;
    Alds[(fl*2+0)*512 + eoff] = (_Float16)Pr[v];
    Alds[(fl*2+1)*512 + eoff] = (_Float16)Pi[v];
    const int fl2 = (8 - u) * 8 + ((8 - v) & 7);
    Alds[(fl2*2+0)*512 + eoff] = (_Float16)Pr[v];
    Alds[(fl2*2+1)*512 + eoff] = (_Float16)(-Pi[v]);
  }
}

// ---------------------------------------------------------------------------
// Fused kernel: patch gather -> factored 8x8 DFT (once per patch) ->
// 64-freq complex GEMM (WMMA f16) -> iFFT as WMMA GEMM vs T -> bias+crop+store.
// grid = (196 tile-blocks, 8 oc-blocks of 32), block = 512 (16 waves).
// Wave owns fl = wave*4+ff (4 freqs) x 2 n-tiles: 16 v8f accumulators.
// ---------------------------------------------------------------------------
__global__ __launch_bounds__(512) void fftconv_main(const float* __restrict__ x,
                                                    const _Float16* __restrict__ wpk,
                                                    const _Float16* __restrict__ wt,
                                                    const float* __restrict__ bias,
                                                    float* __restrict__ out) {
  __shared__ __align__(32) _Float16 Alds[64 * 2 * 512];  // 128 KB, fwd A / iFFT A

  const int tid  = threadIdx.x;
  const int lane = tid & 31;
  const int wave = tid >> 5;
  const int tb   = blockIdx.x;        // tiles tb*16 .. tb*16+15
  const int nt0  = blockIdx.y * 2;    // 2 oc-tiles of 16
  const int oc0  = nt0 * 16;

  // this thread's (tile, ic-within-chunk) pair for the DFT stage
  const int tl  = tid >> 5;           // 0..15
  const int icl = tid & 31;           // 0..31
  const int tg  = tb * 16 + tl;
  const int bi_ = tg / 784;
  const int rr  = (tg / 28) % 28;
  const int cc  = tg % 28;
  // A-operand slot (16-bit A 16x32 layout)
  const int khalf = (icl >> 4) & 1;
  const int avg   = (icl & 15) >> 1;
  const int eoff  = (tl + 16 * khalf) * 16 + avg * 2 + (icl & 1);

  v8f cr[4][2] = {};
  v8f ci[4][2] = {};

  for (int kc = 0; kc < 4; ++kc) {           // IC chunk of 32 (= WMMA K)
    if (kc) __syncthreads();                 // A region free

    // ---- stage 1: gather 6x6 patch, factored 2D DFT, all 64 freqs ----
    {
      const int ic = kc * 32 + icl;
      float p[6][6];
      #pragma unroll
      for (int py = 0; py < 6; ++py) {
        const int yy = rr * 4 + py - 1;
        #pragma unroll
        for (int px = 0; px < 6; ++px) {
          const int xx = cc * 4 + px - 1;
          float v = 0.f;
          if (yy >= 0 && yy < HH && xx >= 0 && xx < WW)
            v = x[(((size_t)bi_ * HH + yy) * WW + xx) * ICN + ic];
          p[py][px] = v;
        }
      }
      // row stage: G[u][px], u=0..4 (G[8-u]=conj(G[u]) handled in colDFT_c)
      float G0[6], G4[6], G1r[6], G1i[6], G2r[6], G2i[6], G3r[6], G3i[6];
      #pragma unroll
      for (int px = 0; px < 6; ++px) {
        const float p0=p[0][px], p1=p[1][px], p2=p[2][px];
        const float p3=p[3][px], p4=p[4][px], p5=p[5][px];
        const float S=p0+p1+p2+p3+p4+p5, T=p0-p1+p2-p3+p4-p5;
        const float D=p0-p2+p4, E=p1-p3+p5;
        const float A=p0-p4, B=p1-p3-p5, C=p1+p3-p5;
        const float rB=RSQ2*B, rC=RSQ2*C;
        G0[px]=S; G4[px]=T;
        G1r[px]=A+rB; G1i[px]=-(p2+rC);
        G2r[px]=D;    G2i[px]=-E;
        G3r[px]=A-rB; G3i[px]=p2-rC;
      }
      colDFT_r(0, G0, Alds, eoff);
      colDFT_r(4, G4, Alds, eoff);
      colDFT_c(1, G1r, G1i, Alds, eoff);
      colDFT_c(2, G2r, G2i, Alds, eoff);
      colDFT_c(3, G3r, G3i, Alds, eoff);
    }
    __syncthreads();

    // ---- stage 2: complex GEMM, wave owns fl = wave*4 + {0..3} ----
    #pragma unroll
    for (int ff = 0; ff < 4; ++ff) {
      const int fl = wave * 4 + ff;
      v16h ar = *reinterpret_cast<const v16h*>(&Alds[(fl*2+0)*512 + lane*16]);
      v16h ai = *reinterpret_cast<const v16h*>(&Alds[(fl*2+1)*512 + lane*16]);
      union { v16h h; unsigned u[8]; } nu;
      nu.h = ai;
      #pragma unroll
      for (int q = 0; q < 8; ++q) nu.u[q] ^= 0x80008000u;  // -Ai
      const v16h nai = nu.h;

      #pragma unroll
      for (int ntl = 0; ntl < 2; ++ntl) {
        const size_t boff =
            ((((size_t)fl * 2 + 0) * 4 + kc) * 16 + (nt0 + ntl)) * 512 + lane * 16;
        v16h br = *reinterpret_cast<const v16h*>(wpk + boff);
        v16h bi = *reinterpret_cast<const v16h*>(wpk + boff + 32768);
        cr[ff][ntl] = __builtin_amdgcn_wmma_f32_16x16x32_f16(false, ar,  false, br, (short)0, cr[ff][ntl], false, false);
        cr[ff][ntl] = __builtin_amdgcn_wmma_f32_16x16x32_f16(false, nai, false, bi, (short)0, cr[ff][ntl], false, false);
        ci[ff][ntl] = __builtin_amdgcn_wmma_f32_16x16x32_f16(false, ar,  false, bi, (short)0, ci[ff][ntl], false, false);
        ci[ff][ntl] = __builtin_amdgcn_wmma_f32_16x16x32_f16(false, ai,  false, br, (short)0, ci[ff][ntl], false, false);
      }
    }
  }

  // ---- stage 3: iFFT as GEMM  O[(m,oc), s] = OF[(m,oc), k] * T[k, s] ----
  // K = 128 = (fl,ri); wave w holds k = w*8 .. w*8+7 -> K-chunk kq = w>>2.
  // iFFT A tiles: tile = m*2 + ntl (32 tiles), row-in-tile = ocl&15.
  v8f co[2] = {};
  for (int kq = 0; kq < 4; ++kq) {
    __syncthreads();                        // A region free
    if ((wave >> 2) == kq) {                // wave-uniform staging group
      const int n_oc = lane & 15;
      const int mb   = (lane >> 4) << 3;    // C: VGPR j -> M = j + 8*(lane>=16)
      const int lane2 = n_oc + 16 * ((wave & 3) >> 1);
      const int vgb   = (wave & 1) * 4;
      #pragma unroll
      for (int ff = 0; ff < 4; ++ff) {
        const int base2 = lane2 * 16 + (vgb + ff) * 2;
        #pragma unroll
        for (int ntl = 0; ntl < 2; ++ntl) {
          #pragma unroll
          for (int j = 0; j < 8; ++j) {
            const int tile = (mb + j) * 2 + ntl;
            Alds[tile * 512 + base2 + 0] = (_Float16)cr[ff][ntl][j];
            Alds[tile * 512 + base2 + 1] = (_Float16)ci[ff][ntl][j];
          }
        }
      }
    }
    __syncthreads();
    v16h bt = *reinterpret_cast<const v16h*>(wt + kq * 512 + lane * 16);
    #pragma unroll
    for (int tt = 0; tt < 2; ++tt) {
      const int tile = wave * 2 + tt;
      v16h at = *reinterpret_cast<const v16h*>(&Alds[tile * 512 + lane * 16]);
      co[tt] = __builtin_amdgcn_wmma_f32_16x16x32_f16(false, at, false, bt, (short)0, co[tt], false, false);
    }
  }

  // ---- stage 4: 1/64 scale, bias, crop, store (C cols = s, rows = ocl&15) ----
  {
    const int s  = lane & 15;
    const int mb = (lane >> 4) << 3;
    #pragma unroll
    for (int tt = 0; tt < 2; ++tt) {
      const int tile = wave * 2 + tt;
      const int m   = tile >> 1;
      const int ntl = tile & 1;
      const int tg2 = tb * 16 + m;
      const int bb  = tg2 / 784;
      const int r2  = (tg2 / 28) % 28;
      const int c2  = tg2 % 28;
      const int I = r2 * 4 + (s >> 2);
      const int J = c2 * 4 + (s & 3);
      if (I >= 2 && J >= 2) {
        const size_t ob = (((size_t)bb * 110 + (I - 2)) * 110 + (J - 2)) * OCN
                          + oc0 + ntl * 16;
        #pragma unroll
        for (int j = 0; j < 8; ++j)
          out[ob + mb + j] = co[tt][j] * 0.015625f + bias[oc0 + ntl * 16 + mb + j];
      }
    }
  }
}

// ---------------------------------------------------------------------------
extern "C" void kernel_launch(void* const* d_in, const int* in_sizes, int n_in,
                              void* d_out, int out_size, void* d_ws, size_t ws_size,
                              hipStream_t stream) {
  (void)in_sizes; (void)n_in; (void)out_size; (void)ws_size;
  const float* x  = (const float*)d_in[0];
  const float* wr = (const float*)d_in[1];   // pre-masked real weights (IC,OC,8,8)
  const float* wi = (const float*)d_in[2];   // pre-masked imag weights
  const float* bs = (const float*)d_in[4];   // bias (OC) ; d_in[3] mask unused
  float* out = (float*)d_out;
  _Float16* wpk = (_Float16*)d_ws;                                  // 8 MB B tiles
  _Float16* wt  = (_Float16*)((char*)d_ws + (8u << 20));            // 4 KB T tiles

  pack_w<<<16384, 256, 0, stream>>>(wr, wi, wpk);
  pack_t<<<8, 256, 0, stream>>>(wt);
  fftconv_main<<<dim3(196, 8), 512, 0, stream>>>(x, wpk, wt, bs, out);
}
